// GAT_48129403519230
// MI455X (gfx1250) — matrix operand order repaired and checked by reading, not compiled, not run on hardware
//
#include <hip/hip_runtime.h>
#include <math.h>

// ---------------- problem constants (match reference) ----------------
#define NNODES 50000
#define NEDGES 400000
#define INCH   128
#define NH     4
#define CH     64
#define HC     256          // NH*CH
#define NQ     200000
#define NEG_SLOPE 0.2f
#define BN_EPS    1e-5f
#define EADJ   (NEDGES + NNODES)   // edges incl. self loops

typedef __attribute__((ext_vector_type(16))) _Float16 v16h;
typedef __attribute__((ext_vector_type(8)))  float    v8f;
typedef __attribute__((ext_vector_type(4)))  _Float16 h4;
typedef __attribute__((ext_vector_type(2)))  _Float16 h2;

#define LPITCH 40   // LDS row pitch in halves: 80 B, 16B-aligned, bank-spread

// ---------------- order-preserving float<->uint for atomicMax ----------------
__device__ __forceinline__ unsigned f2ord(float f) {
    unsigned u = __float_as_uint(f);
    return (u & 0x80000000u) ? ~u : (u | 0x80000000u);
}
__device__ __forceinline__ float ord2f(unsigned u) {
    return __uint_as_float((u & 0x80000000u) ? (u ^ 0x80000000u) : ~u);
}

// =====================================================================
// WMMA GEMM: Out[M x Nc] = A[M x K] (f32) * W[K x Nc] (f32) + bias[Nc]
// f32 -> f16 staging in LDS (vectorized b128 global loads, packed LDS
// stores), f32 accumulation via v_wmma_f32_16x16x32_f16.
//
// LDS layouts are chosen so each lane's 16 fragment halves are CONTIGUOUS:
//  - A[row][pos]: pos = k with bits 3 and 4 swapped. Then the v16h A-frag
//    for (row, half) is lA[row][16*half .. +15]  (ISA 7.12.2 order).
//  - B staged transposed lBt[col][apos(k)]: v16h B-frag for (col, half) is
//    lBt[col][16*half .. +15].
// Block = 128 threads (4 waves). Block tile 64 rows x 64 cols.
// =====================================================================
__device__ __forceinline__ int apos(int k) {       // swap bits 3 and 4
    return (k & 7) | ((k & 8) << 1) | ((k & 16) >> 1);
}

__device__ __forceinline__ void storeCfrag(v8f c, float* __restrict__ Out, int Nc, int M,
                                           int rowTop, int col, float bv) {
    // C/D 16x16 f32 layout: lane half h, reg r -> M = 8h + r, N = lane%16
#pragma unroll
    for (int r = 0; r < 8; ++r) {
        int grow = rowTop + r;
        if (grow < M) Out[(size_t)grow * Nc + col] = c[r] + bv;
    }
}

__global__ __launch_bounds__(128)
void gemm_wmma_bias(const float* __restrict__ A, const float* __restrict__ W,
                    const float* __restrict__ bias, float* __restrict__ Out,
                    int M, int K, int Nc)
{
    __shared__ _Float16 lA[64][LPITCH];    // 64 rows x 32 k (k-permuted)
    __shared__ _Float16 lBt[64][LPITCH];   // 64 cols x 32 k (transposed)

    const int tid  = threadIdx.x;
    const int lane = tid & 31;
    const int wv   = tid >> 5;
    const int rowBase = blockIdx.x * 64;
    const int colBase = blockIdx.y * 64;
    const int half = lane >> 4;     // 0 or 1
    const int l16  = lane & 15;
    const bool fullTile = (rowBase + 64 <= M);

    v8f acc0 = {}; v8f acc1 = {}; v8f acc2 = {}; v8f acc3 = {};

    for (int k0 = 0; k0 < K; k0 += 32) {
        // ---- stage A: 64 rows x 32 k, float4 granularity (8 float4 / row) ----
#pragma unroll
        for (int l = 0; l < 4; ++l) {
            int i   = tid + l * 128;       // 0..511
            int r   = i >> 3;
            int k   = (i & 7) * 4;
            float4 v = make_float4(0.f, 0.f, 0.f, 0.f);
            int gr = rowBase + r;
            if (fullTile || gr < M)
                v = *(const float4*)(A + (size_t)gr * K + (k0 + k));
            h4 hv;
            hv[0] = (_Float16)v.x; hv[1] = (_Float16)v.y;
            hv[2] = (_Float16)v.z; hv[3] = (_Float16)v.w;
            *(h4*)&lA[r][apos(k)] = hv;    // apos(k) multiple of 4 -> 8B aligned
        }
        // ---- stage B transposed: lBt[c][apos(k)], packed half2 along k ----
#pragma unroll
        for (int l = 0; l < 8; ++l) {
            int i  = tid + l * 128;        // 0..1023
            int c  = i & 63;
            int k  = (i >> 6) * 2;
            float f0 = W[(size_t)(k0 + k)     * Nc + (colBase + c)];
            float f1 = W[(size_t)(k0 + k + 1) * Nc + (colBase + c)];
            h2 hv; hv[0] = (_Float16)f0; hv[1] = (_Float16)f1;
            *(h2*)&lBt[c][apos(k)] = hv;   // k even -> 4B aligned
        }
        __syncthreads();

        // ---- contiguous 32B fragment loads (2x ds_load_b128 each) ----
        v16h af = *(const v16h*)&lA[wv * 16 + l16][half * 16];
        v16h b0 = *(const v16h*)&lBt[ 0 + l16][half * 16];
        v16h b1 = *(const v16h*)&lBt[16 + l16][half * 16];
        v16h b2 = *(const v16h*)&lBt[32 + l16][half * 16];
        v16h b3 = *(const v16h*)&lBt[48 + l16][half * 16];

        acc0 = __builtin_amdgcn_wmma_f32_16x16x32_f16(false, af, false, b0, (short)0, acc0, false, false);
        acc1 = __builtin_amdgcn_wmma_f32_16x16x32_f16(false, af, false, b1, (short)0, acc1, false, false);
        acc2 = __builtin_amdgcn_wmma_f32_16x16x32_f16(false, af, false, b2, (short)0, acc2, false, false);
        acc3 = __builtin_amdgcn_wmma_f32_16x16x32_f16(false, af, false, b3, (short)0, acc3, false, false);
        __syncthreads();
    }

    const int rowTop = rowBase + wv * 16 + half * 8;
    storeCfrag(acc0, Out, Nc, M, rowTop, colBase +  0 + l16, bias[colBase +  0 + l16]);
    storeCfrag(acc1, Out, Nc, M, rowTop, colBase + 16 + l16, bias[colBase + 16 + l16]);
    storeCfrag(acc2, Out, Nc, M, rowTop, colBase + 32 + l16, bias[colBase + 32 + l16]);
    storeCfrag(acc3, Out, Nc, M, rowTop, colBase + 48 + l16, bias[colBase + 48 + l16]);
}

// =====================================================================
// Edge phase kernels: one wave32 per (edge, head); lanes cover the 64 channels.
// Self loops implicit: e >= NEDGES -> src = dst = e - NEDGES.
// =====================================================================
__device__ __forceinline__ void edge_src_dst(const int* __restrict__ ei, int e, int& src, int& dst) {
    if (e < NEDGES) { src = ei[e]; dst = ei[NEDGES + e]; }
    else            { src = dst = e - NEDGES; }
}

__global__ __launch_bounds__(256)
void edge_logits_max(const int* __restrict__ ei,
                     const float* __restrict__ xl, const float* __restrict__ xr,
                     const float* __restrict__ att,
                     float* __restrict__ logits, unsigned* __restrict__ mEnc)
{
    int gw   = (blockIdx.x * 256 + threadIdx.x) >> 5;    // global wave id
    int lane = threadIdx.x & 31;
    if (gw >= EADJ * NH) return;
    int e = gw >> 2, h = gw & 3;
    int src, dst; edge_src_dst(ei, e, src, dst);

    const float* pl = xl + (size_t)src * HC + h * CH;
    const float* pr = xr + (size_t)dst * HC + h * CH;
    const float* pa = att + h * CH;

    float acc = 0.0f;
#pragma unroll
    for (int c = lane; c < CH; c += 32) {
        float v = pl[c] + pr[c];
        v = (v > 0.0f) ? v : NEG_SLOPE * v;
        acc += pa[c] * v;
    }
#pragma unroll
    for (int off = 16; off > 0; off >>= 1) acc += __shfl_xor(acc, off, 32);

    if (lane == 0) {
        logits[(size_t)e * NH + h] = acc;
        atomicMax(&mEnc[(size_t)dst * NH + h], f2ord(acc));
    }
}

__global__ __launch_bounds__(256)
void edge_exp_sum(const int* __restrict__ ei,
                  float* __restrict__ p,            // in: logits, out: exp(logit - m)
                  const unsigned* __restrict__ mEnc,
                  float* __restrict__ s)
{
    int t = blockIdx.x * 256 + threadIdx.x;
    if (t >= EADJ * NH) return;
    int e = t >> 2, h = t & 3;
    int src, dst; edge_src_dst(ei, e, src, dst); (void)src;
    float m = ord2f(mEnc[(size_t)dst * NH + h]);
    float v = expf(p[t] - m);
    p[t] = v;
    atomicAdd(&s[(size_t)dst * NH + h], v);
}

__global__ __launch_bounds__(256)
void edge_scatter(const int* __restrict__ ei,
                  const float* __restrict__ xl,
                  const float* __restrict__ p, const float* __restrict__ s,
                  float* __restrict__ acc)
{
    int gw   = (blockIdx.x * 256 + threadIdx.x) >> 5;
    int lane = threadIdx.x & 31;
    if (gw >= EADJ * NH) return;
    int e = gw >> 2, h = gw & 3;
    int src, dst; edge_src_dst(ei, e, src, dst);

    float alpha = p[(size_t)e * NH + h] / s[(size_t)dst * NH + h];
    const float* pl = xl  + (size_t)src * HC + h * CH;
    float*       po = acc + (size_t)dst * HC + h * CH;
#pragma unroll
    for (int c = lane; c < CH; c += 32)
        atomicAdd(&po[c], pl[c] * alpha);
}

__global__ __launch_bounds__(256)
void head_mean_bias(const float* __restrict__ acc, const float* __restrict__ bias,
                    float* __restrict__ out)
{
    int t = blockIdx.x * 256 + threadIdx.x;
    if (t >= NNODES * CH) return;
    int n = t >> 6, c = t & 63;
    const float* a = acc + (size_t)n * HC;
    out[t] = 0.25f * (a[c] + a[CH + c] + a[2 * CH + c] + a[3 * CH + c]) + bias[c];
}

// =====================================================================
// BatchNorm stats (biased var, like jnp.var): one block per column.
// stats[c] = mean, stats[64+c] = rsqrt(var + eps)
// =====================================================================
__global__ __launch_bounds__(256)
void bn_stats(const float* __restrict__ xx, float* __restrict__ stats)
{
    __shared__ float sh[256], sh2[256];
    int col = blockIdx.x;
    float s = 0.0f, s2 = 0.0f;
    for (int n = threadIdx.x; n < NNODES; n += 256) {
        float v = xx[(size_t)n * CH + col];
        s += v; s2 += v * v;
    }
    sh[threadIdx.x] = s; sh2[threadIdx.x] = s2;
    __syncthreads();
    for (int st = 128; st > 0; st >>= 1) {
        if (threadIdx.x < st) { sh[threadIdx.x] += sh[threadIdx.x + st]; sh2[threadIdx.x] += sh2[threadIdx.x + st]; }
        __syncthreads();
    }
    if (threadIdx.x == 0) {
        float mu  = sh[0] / (float)NNODES;
        float var = sh2[0] / (float)NNODES - mu * mu;
        stats[col]      = mu;
        stats[CH + col] = rsqrtf(var + BN_EPS);
    }
}

__global__ __launch_bounds__(256)
void post1_kernel(const float* __restrict__ gat, const float* __restrict__ stats,
                  const float* __restrict__ g, const float* __restrict__ b,
                  const float* __restrict__ xres, float* __restrict__ h1)
{
    int t = blockIdx.x * 256 + threadIdx.x;
    if (t >= NNODES * CH) return;
    int c = t & 63;
    float v = g[c] * (gat[t] - stats[c]) * stats[CH + c] + b[c] + xres[t];
    h1[t] = fmaxf(v, 0.0f);
}

__global__ __launch_bounds__(256)
void post2_kernel(const float* __restrict__ gat, const float* __restrict__ stats,
                  const float* __restrict__ g, const float* __restrict__ b,
                  const float* __restrict__ h1, float* __restrict__ z)
{
    int t = blockIdx.x * 256 + threadIdx.x;
    if (t >= NNODES * CH) return;
    int c = t & 63;
    z[t] = g[c] * (gat[t] - stats[c]) * stats[CH + c] + b[c] + h1[t];
}

// =====================================================================
// Link-prediction scores: one wave per query, lanes over 64 channels.
// =====================================================================
__global__ __launch_bounds__(256)
void score_kernel(const int* __restrict__ eli, const float* __restrict__ z,
                  float* __restrict__ out)
{
    int gw   = (blockIdx.x * 256 + threadIdx.x) >> 5;
    int lane = threadIdx.x & 31;
    if (gw >= NQ) return;
    int a = eli[gw], bidx = eli[NQ + gw];
    const float* za = z + (size_t)a * CH;
    const float* zb = z + (size_t)bidx * CH;
    float acc = za[lane] * zb[lane] + za[lane + 32] * zb[lane + 32];
#pragma unroll
    for (int off = 16; off > 0; off >>= 1) acc += __shfl_xor(acc, off, 32);
    if (lane == 0) out[gw] = 1.0f / (1.0f + expf(-acc));
}

// =====================================================================
extern "C" void kernel_launch(void* const* d_in, const int* in_sizes, int n_in,
                              void* d_out, int out_size, void* d_ws, size_t ws_size,
                              hipStream_t stream)
{
    (void)in_sizes; (void)n_in; (void)out_size; (void)ws_size;
    const float* x     = (const float*)d_in[0];
    const int*   ei    = (const int*)d_in[1];
    const int*   eli   = (const int*)d_in[2];
    const float* Wl1   = (const float*)d_in[3];
    const float* bl1   = (const float*)d_in[4];
    const float* Wr1   = (const float*)d_in[5];
    const float* br1   = (const float*)d_in[6];
    const float* att1  = (const float*)d_in[7];
    const float* bias1 = (const float*)d_in[8];
    const float* g1    = (const float*)d_in[9];
    const float* b1    = (const float*)d_in[10];
    const float* Wres  = (const float*)d_in[11];
    const float* bres  = (const float*)d_in[12];
    const float* Wl2   = (const float*)d_in[13];
    const float* bl2   = (const float*)d_in[14];
    const float* Wr2   = (const float*)d_in[15];
    const float* br2   = (const float*)d_in[16];
    const float* att2  = (const float*)d_in[17];
    const float* bias2 = (const float*)d_in[18];
    const float* g2    = (const float*)d_in[19];
    const float* b2    = (const float*)d_in[20];
    float* out = (float*)d_out;

    // ---- workspace carve-out (256B aligned), ~227 MB total ----
    char* wp = (char*)d_ws;
    auto take = [&](size_t bytes) -> char* {
        char* p = wp; wp += (bytes + 255) & ~(size_t)255; return p;
    };
    float*    XL    = (float*)take((size_t)NNODES * HC * 4);  // xl projection (L2 resident)
    float*    XR    = (float*)take((size_t)NNODES * HC * 4);  // xr projection (L2 resident)
    float*    ACC   = (float*)take((size_t)NNODES * HC * 4);  // per-head scatter accumulator
    float*    XRES  = (float*)take((size_t)NNODES * CH * 4);
    float*    H1    = (float*)take((size_t)NNODES * CH * 4);
    float*    G1O   = (float*)take((size_t)NNODES * CH * 4);
    float*    G2O   = (float*)take((size_t)NNODES * CH * 4);
    float*    Z     = (float*)take((size_t)NNODES * CH * 4);
    float*    P     = (float*)take((size_t)EADJ * NH * 4);    // logits -> softmax numerators
    unsigned* MENC  = (unsigned*)take((size_t)NNODES * NH * 4);
    float*    S     = (float*)take((size_t)NNODES * NH * 4);
    float*    STATS = (float*)take(2 * CH * 4);

    dim3 gemmBlk(128);
    dim3 gGemm256((NNODES + 63) / 64, HC / 64);
    dim3 gGemm64((NNODES + 63) / 64, 1);

    const int nWaves = EADJ * NH;
    dim3 eblk(256);
    dim3 egrid((nWaves + 7) / 8);                  // 8 waves / block
    dim3 tgrid((nWaves + 255) / 256);              // thread-per-(e,h)
    dim3 ncgrid((NNODES * CH + 255) / 256);        // thread-per-(n,c)
    dim3 sgrid((NQ + 7) / 8);                      // wave-per-query

    // ================= layer 1 =================
    gemm_wmma_bias<<<gGemm256, gemmBlk, 0, stream>>>(x, Wl1, bl1, XL, NNODES, INCH, HC);
    gemm_wmma_bias<<<gGemm256, gemmBlk, 0, stream>>>(x, Wr1, br1, XR, NNODES, INCH, HC);
    gemm_wmma_bias<<<gGemm64,  gemmBlk, 0, stream>>>(x, Wres, bres, XRES, NNODES, INCH, CH);

    hipMemsetAsync(MENC, 0, (size_t)NNODES * NH * 4, stream);
    hipMemsetAsync(S,    0, (size_t)NNODES * NH * 4, stream);
    hipMemsetAsync(ACC,  0, (size_t)NNODES * HC * 4, stream);

    edge_logits_max<<<egrid, eblk, 0, stream>>>(ei, XL, XR, att1, P, MENC);
    edge_exp_sum  <<<tgrid, eblk, 0, stream>>>(ei, P, MENC, S);
    edge_scatter  <<<egrid, eblk, 0, stream>>>(ei, XL, P, S, ACC);
    head_mean_bias<<<ncgrid, eblk, 0, stream>>>(ACC, bias1, G1O);
    bn_stats      <<<dim3(CH), dim3(256), 0, stream>>>(G1O, STATS);
    post1_kernel  <<<ncgrid, eblk, 0, stream>>>(G1O, STATS, g1, b1, XRES, H1);

    // ================= layer 2 =================
    gemm_wmma_bias<<<gGemm256, gemmBlk, 0, stream>>>(H1, Wl2, bl2, XL, NNODES, CH, HC);
    gemm_wmma_bias<<<gGemm256, gemmBlk, 0, stream>>>(H1, Wr2, br2, XR, NNODES, CH, HC);

    hipMemsetAsync(MENC, 0, (size_t)NNODES * NH * 4, stream);
    hipMemsetAsync(S,    0, (size_t)NNODES * NH * 4, stream);
    hipMemsetAsync(ACC,  0, (size_t)NNODES * HC * 4, stream);

    edge_logits_max<<<egrid, eblk, 0, stream>>>(ei, XL, XR, att2, P, MENC);
    edge_exp_sum  <<<tgrid, eblk, 0, stream>>>(ei, P, MENC, S);
    edge_scatter  <<<egrid, eblk, 0, stream>>>(ei, XL, P, S, ACC);
    head_mean_bias<<<ncgrid, eblk, 0, stream>>>(ACC, bias2, G2O);
    bn_stats      <<<dim3(CH), dim3(256), 0, stream>>>(G2O, STATS);
    post2_kernel  <<<ncgrid, eblk, 0, stream>>>(G2O, STATS, g2, b2, H1, Z);

    // ================= link prediction =================
    score_kernel<<<sgrid, eblk, 0, stream>>>(eli, Z, out);
}
